// DeltaNetModel_80805514707473
// MI455X (gfx1250) — matrix operand order repaired and checked by reading, not compiled, not run on hardware
//
#include <hip/hip_runtime.h>

// ---------------- CDNA5 WMMA types / helpers ----------------
typedef __attribute__((ext_vector_type(16))) _Float16 v16h;
typedef __attribute__((ext_vector_type(8)))  _Float16 v8h;
typedef __attribute__((ext_vector_type(8)))  float    v8f;

__device__ __forceinline__ v8f wmma_f16f32(v16h a, v16h b, v8f c) {
  // D(16x16 f32) = A(16x32 f16) * B(32x16 f16) + C
  return __builtin_amdgcn_wmma_f32_16x16x32_f16(
      false, a, false, b, (short)0, c, false, false);
}

// A-fragment loader: halves j<8 -> k = base+j ; j>=8 -> k = base+16+(j-8)
// (pass p = row + kc + hi*8). Two 16B vector loads.
__device__ __forceinline__ v16h load_a16(const _Float16* p) {
  v8h lo = *(const v8h*)p;
  v8h hh = *(const v8h*)(p + 16);
  v16h a;
#pragma unroll
  for (int j = 0; j < 8; ++j) { a[j] = lo[j]; a[j + 8] = hh[j]; }
  return a;
}

__device__ __forceinline__ float wave_rsum(float v) {
#pragma unroll
  for (int o = 16; o; o >>= 1) v += __shfl_xor(v, o);
  return v;
}
__device__ __forceinline__ float wave_rmax(float v) {
#pragma unroll
  for (int o = 16; o; o >>= 1) v = fmaxf(v, __shfl_xor(v, o));
  return v;
}

#define TT 32
#define BB 16
#define NHH 4
#define DHH 32
#define HIDD 128
#define NAA 18
#define XPITCH 288   // core input row pitch (257 padded to 288)

// ---------------- weight f32 -> f16 with row padding ----------------
__global__ void k_cvtpad(const float* __restrict__ src, _Float16* __restrict__ dst,
                         int rows, int K, int KPAD, int rowsPad) {
  int i = blockIdx.x * blockDim.x + threadIdx.x;
  if (i >= rowsPad * KPAD) return;
  int r = i / KPAD, k = i % KPAD;
  float v = (r < rows && k < K) ? src[r * K + k] : 0.0f;
  dst[i] = (_Float16)v;
}

// ---------------- zero-padded (+1 ring) input stage, optional relu/scale -------
template <typename SRC>
__global__ void k_pad(const SRC* __restrict__ in, _Float16* __restrict__ out,
                      int CI, int CIP, int S, float scale, int relu, int total) {
  int i = blockIdx.x * blockDim.x + threadIdx.x;
  if (i >= total) return;
  int PS = S + 2;
  int x = i % PS; int t = i / PS;
  int y = t % PS; t /= PS;
  int c = t % CIP; int img = t / CIP;
  float v = 0.0f;
  if (c < CI && y >= 1 && y <= S && x >= 1 && x <= S) {
    v = (float)in[(((size_t)img * CI + c) * S + (y - 1)) * S + (x - 1)] * scale;
    if (relu) v = fmaxf(v, 0.0f);
  }
  out[i] = (_Float16)v;
}

// ---------------- conv3x3 pad1 implicit-GEMM WMMA --------------------------
// One A gather feeds NT x 16-column B tiles (register blocking on out-channels).
// pin [512][CIP][S+2][S+2] f16 ; w16 [CO][KPAD] f16 ; out [512][CO][S][S] f16
// grid (S*S/16/TPB, CO/(16*NT), 512), block 32*TPB
template <int CI, int S, int NT>
__global__ void k_conv3x3(const _Float16* __restrict__ pin,
                          const _Float16* __restrict__ w16,
                          const float* __restrict__ bias,
                          const _Float16* __restrict__ resid,
                          _Float16* __restrict__ out, int CO, int relu_out) {
  constexpr int K9 = CI * 9;
  constexpr int KPAD = ((K9 + 31) / 32) * 32;
  constexpr int CIP = (KPAD + 8) / 9;   // zero channels cover K tail
  constexpr int PS = S + 2;
  const int lane = threadIdx.x & 31;
  const int wv = threadIdx.x >> 5;
  const int m = lane & 15, hi = lane >> 4;
  const int tile = blockIdx.x * (blockDim.x >> 5) + wv;
  const int img = blockIdx.z;
  const int pix = tile * 16 + m;
  const int py = pix / S, px = pix % S;
  const _Float16* pbase = pin + ((size_t)img * CIP * PS + py) * PS + px;
  const int n0 = blockIdx.y * (16 * NT) + m;

  const _Float16* wrow[NT];
  v8f acc[NT];
#pragma unroll
  for (int u = 0; u < NT; ++u) {
    wrow[u] = w16 + (size_t)(n0 + u * 16) * KPAD;
    acc[u] = v8f{};
  }
#pragma unroll
  for (int kc = 0; kc < KPAD; kc += 32) {
    v16h a;
#pragma unroll
    for (int j = 0; j < 16; ++j) {
      int ka0 = kc + (j & 7) + ((j >> 3) << 4);   // hi==0 k (constant)
      int ka1 = ka0 + 8;                          // hi==1 k (constant)
      int o0 = ((ka0 / 9) * PS + (ka0 % 9) / 3) * PS + (ka0 % 9) % 3;
      int o1 = ((ka1 / 9) * PS + (ka1 % 9) / 3) * PS + (ka1 % 9) % 3;
      a[j] = pbase[hi ? o1 : o0];
    }
#pragma unroll
    for (int u = 0; u < NT; ++u) {
      v16h b = load_a16(wrow[u] + kc + hi * 8);
      acc[u] = wmma_f16f32(a, b, acc[u]);
    }
  }
#pragma unroll
  for (int u = 0; u < NT; ++u) {
    const int n = n0 + u * 16;
    _Float16*       oimg = out + ((size_t)img * CO + n) * (S * S);
    const _Float16* rimg = resid ? resid + ((size_t)img * CO + n) * (S * S) : nullptr;
    const float bn = bias[n];
#pragma unroll
    for (int r = 0; r < 8; ++r) {
      int p = tile * 16 + r + hi * 8;
      float v = acc[u][r] + bn;
      if (rimg) v += (float)rimg[p];
      if (relu_out) v = fmaxf(v, 0.0f);
      oimg[p] = (_Float16)v;
    }
  }
}

// ---------------- maxpool 3x3 stride2 pad1 ----------------
__global__ void k_maxpool(const _Float16* __restrict__ in, _Float16* __restrict__ out,
                          int C, int S, int total) {
  int idx = blockIdx.x * blockDim.x + threadIdx.x;
  if (idx >= total) return;
  int SO = S >> 1;
  int ox = idx % SO; int t = idx / SO;
  int oy = t % SO;   t /= SO;
  int c  = t % C;    int nimg = t / C;
  const _Float16* p = in + ((size_t)nimg * C + c) * S * S;
  float mx = -3.0e38f;
#pragma unroll
  for (int dy = -1; dy <= 1; ++dy) {
    int yy = 2 * oy + dy;
    if ((unsigned)yy >= (unsigned)S) continue;
#pragma unroll
    for (int dx = -1; dx <= 1; ++dx) {
      int xx = 2 * ox + dx;
      if ((unsigned)xx >= (unsigned)S) continue;
      mx = fmaxf(mx, (float)p[yy * S + xx]);
    }
  }
  out[idx] = (_Float16)mx;
}

// ---------------- FC -> core_in16 [512][288] f16 (cols 0..255) ----------------
// grid (4,16), block 256
__global__ void __launch_bounds__(256)
k_fc(const _Float16* __restrict__ act, const _Float16* __restrict__ fw16,
     const float* __restrict__ fb, _Float16* __restrict__ core_in16) {
  const int lane = threadIdx.x & 31, wv = threadIdx.x >> 5;
  const int m = lane & 15, hi = lane >> 4;
  const int m0 = (blockIdx.x * 8 + wv) * 16, n0 = blockIdx.y * 16;
  const _Float16* arow = act + (size_t)(m0 + m) * 2048;
  const _Float16* wrow = fw16 + (size_t)(n0 + m) * 2048;
  v8f acc = {};
  for (int kc = 0; kc < 2048; kc += 32) {
    __builtin_prefetch(arow + kc + 256, 0, 0);
    v16h a = load_a16(arow + kc + hi * 8);
    v16h b = load_a16(wrow + kc + hi * 8);
    acc = wmma_f16f32(a, b, acc);
  }
  int n = n0 + m;
  float bn = fb[n];
#pragma unroll
  for (int r = 0; r < 8; ++r)
    core_in16[(size_t)(m0 + r + hi * 8) * XPITCH + n] =
        (_Float16)fmaxf(acc[r] + bn, 0.0f);
}

// ---------------- tail columns: col 256 = clipped reward, 257..287 = 0 --------
__global__ void k_tailcols(const float* __restrict__ reward,
                           _Float16* __restrict__ core_in16) {
  int i = blockIdx.x * blockDim.x + threadIdx.x;   // 512 * 32
  if (i >= TT * BB * 32) return;
  int row = i >> 5, c = 256 + (i & 31);
  float v = 0.0f;
  if (c == 256) v = fminf(fmaxf(reward[row], -1.0f), 1.0f);
  core_in16[(size_t)row * XPITCH + c] = (_Float16)v;
}

// ---------------- recurrent DeltaNet core ----------------
struct CoreLayer {
  const float *ln_g, *ln_b, *ffln_g, *ffln_b, *ff1_b, *ff2_b;
  const _Float16 *slow_w, *out_w, *ff1_w, *ff2_w;   // f16, padded rows
};

__device__ __forceinline__ void ln_rows(const float* __restrict__ hs,
                                        const float* __restrict__ g,
                                        const float* __restrict__ bt,
                                        _Float16* __restrict__ ys, int wv, int lane) {
#pragma unroll
  for (int rr = 0; rr < 2; ++rr) {
    int b = wv * 2 + rr;
    const float* row = hs + b * HIDD;
    float4 x4 = *(const float4*)(row + lane * 4);
    float s  = x4.x + x4.y + x4.z + x4.w;
    float s2 = x4.x * x4.x + x4.y * x4.y + x4.z * x4.z + x4.w * x4.w;
    s = wave_rsum(s); s2 = wave_rsum(s2);
    float mean = s * (1.0f / HIDD);
    float inv  = rsqrtf(s2 * (1.0f / HIDD) - mean * mean + 1e-5f);
#pragma unroll
    for (int j = 0; j < 4; ++j) {
      int c = lane * 4 + j;
      float xv = (j == 0) ? x4.x : (j == 1) ? x4.y : (j == 2) ? x4.z : x4.w;
      ys[b * HIDD + c] = (_Float16)((xv - mean) * inv * g[c] + bt[c]);
    }
  }
}

// M=16 GEMM tile: A = f16 LDS [16][apitch], B = f16 global padded [*][kpad]
__device__ __forceinline__ v8f gemm16(const _Float16* __restrict__ As, int apitch,
                                      const _Float16* __restrict__ W16, int kpad,
                                      int n, int lane) {
  const int m = lane & 15, hi = lane >> 4;
  const _Float16* arow = As + m * apitch;
  const _Float16* wrow = W16 + (size_t)n * kpad;
  v8f acc = {};
#pragma unroll
  for (int kc = 0; kc < kpad; kc += 32) {
    v16h a = load_a16(arow + kc + hi * 8);
    v16h b = load_a16(wrow + kc + hi * 8);
    acc = wmma_f16f32(a, b, acc);
  }
  return acc;
}

__global__ void __launch_bounds__(256)
k_core(const _Float16* __restrict__ core_in16, const int* __restrict__ done,
       const _Float16* __restrict__ inp_w16, const float* __restrict__ inp_b,
       CoreLayer L0, CoreLayer L1,
       float* __restrict__ s0, float* __restrict__ s1,
       float* __restrict__ outs) {
  __shared__ __attribute__((aligned(16))) _Float16 xs[16 * XPITCH]; // x_t f16
  __shared__ float    hs[16 * HIDD];  // hidden f32
  __shared__ _Float16 ys[16 * HIDD];  // LN output / fast-weight output o
  __shared__ float    scr[16 * 512];  // qkvb slots [b][h][128] ; f16 ff overlay
  __shared__ float    nd_s[BB];

  const int tid  = threadIdx.x;
  const int lane = tid & 31, wv = tid >> 5;
  const int m    = lane & 15, hi = lane >> 4;

  for (int t = 0; t < TT; ++t) {
    if (tid < BB) nd_s[tid] = done[t * BB + tid] ? 0.0f : 1.0f;
    // stage x_t (16 rows x 288 f16 = 9216B, contiguous) via CDNA5 async DMA to LDS
    {
      const _Float16* src = core_in16 + (size_t)t * BB * XPITCH;
      const unsigned lds0 = (unsigned)(size_t)(&xs[0]);     // LDS byte offset
      for (int i = tid; i < (BB * XPITCH * 2) / 16; i += 256) {   // 576 x b128
        unsigned dst = lds0 + i * 16;
        unsigned long long sa = (unsigned long long)(size_t)src + (unsigned long long)i * 16;
        asm volatile("global_load_async_to_lds_b128 %0, %1, off"
                     :: "v"(dst), "v"(sa) : "memory");
      }
      asm volatile("s_wait_asynccnt 0" ::: "memory");
    }
    __syncthreads();
    // episode-boundary state reset (b128)
    for (int i = tid * 4; i < BB * NHH * DHH * DHH; i += 256 * 4) {
      float nd = nd_s[i >> 12];
      float4 a = *(float4*)(s0 + i); float4 b = *(float4*)(s1 + i);
      a.x *= nd; a.y *= nd; a.z *= nd; a.w *= nd;
      b.x *= nd; b.y *= nd; b.z *= nd; b.w *= nd;
      *(float4*)(s0 + i) = a; *(float4*)(s1 + i) = b;
    }
    // h = x @ inp_w.T + inp_b
    {
      int n = wv * 16 + m;
      v8f acc = gemm16(xs, XPITCH, inp_w16, XPITCH, n, lane);
      float bn = inp_b[n];
#pragma unroll
      for (int r = 0; r < 8; ++r) hs[(r + hi * 8) * HIDD + n] = acc[r] + bn;
    }
    __syncthreads();

    for (int layer = 0; layer < 2; ++layer) {
      const CoreLayer& L = layer ? L1 : L0;
      float* W = layer ? s1 : s0;

      // ---- fastff ----
      ln_rows(hs, L.ln_g, L.ln_b, ys, wv, lane);
      __syncthreads();
      for (int tile = wv; tile < 25; tile += 8) {
        int n = tile * 16 + m;
        v8f acc = gemm16(ys, HIDD, L.slow_w, HIDD, n, lane);
        if (n < 388) {
          int hd = n / 97, wi = n % 97;   // slot [b][h][128] = q32|k32|v32|beta
#pragma unroll
          for (int r = 0; r < 8; ++r)
            scr[((r + hi * 8) * NHH + hd) * 128 + wi] = acc[r];
        }
      }
      __syncthreads();
      for (int pp = 0; pp < 8; ++pp) {
        int pair = wv * 8 + pp;
        int b = pair >> 2, h = pair & 3;
        float* slot = scr + (b * NHH + h) * 128;
        float q = slot[lane];
        q = __expf(q - wave_rmax(q)); q = q / wave_rsum(q);
        float k = slot[32 + lane];
        k = __expf(k - wave_rmax(k)); k = k / wave_rsum(k);
        float bt = 1.0f / (1.0f + __expf(-slot[96]));
        float vv = slot[64 + lane];
        slot[lane] = q;
        slot[32 + lane] = k;
        float4* w4 = (float4*)(W + ((size_t)(b * NHH + h) * DHH + lane) * DHH);
        float4 wr[8];
        float vhat = 0.0f;
#pragma unroll
        for (int d4 = 0; d4 < 8; ++d4) {
          wr[d4] = w4[d4];
          float4 kk = *(const float4*)(slot + 32 + d4 * 4);
          vhat += wr[d4].x * kk.x + wr[d4].y * kk.y + wr[d4].z * kk.z + wr[d4].w * kk.w;
        }
        float coef = bt * (vv - vhat);
        float ov = 0.0f;
#pragma unroll
        for (int d4 = 0; d4 < 8; ++d4) {
          float4 kk = *(const float4*)(slot + 32 + d4 * 4);
          float4 qq = *(const float4*)(slot + d4 * 4);
          wr[d4].x += coef * kk.x; wr[d4].y += coef * kk.y;
          wr[d4].z += coef * kk.z; wr[d4].w += coef * kk.w;
          ov += wr[d4].x * qq.x + wr[d4].y * qq.y + wr[d4].z * qq.z + wr[d4].w * qq.w;
          w4[d4] = wr[d4];
        }
        ys[b * HIDD + h * DHH + lane] = (_Float16)ov;
      }
      __syncthreads();
      {
        int n = wv * 16 + m;
        v8f acc = gemm16(ys, HIDD, L.out_w, HIDD, n, lane);
#pragma unroll
        for (int r = 0; r < 8; ++r) hs[(r + hi * 8) * HIDD + n] += acc[r];
      }
      __syncthreads();

      // ---- ff block ----
      ln_rows(hs, L.ffln_g, L.ffln_b, ys, wv, lane);
      __syncthreads();
      _Float16* us = (_Float16*)scr;   // [16][512] f16 overlay
      for (int tile = wv; tile < 32; tile += 8) {
        int n = tile * 16 + m;
        v8f acc = gemm16(ys, HIDD, L.ff1_w, HIDD, n, lane);
        float bn = L.ff1_b[n];
#pragma unroll
        for (int r = 0; r < 8; ++r)
          us[(r + hi * 8) * 512 + n] = (_Float16)fmaxf(acc[r] + bn, 0.0f);
      }
      __syncthreads();
      {
        int n = wv * 16 + m;
        v8f acc = gemm16(us, 512, L.ff2_w, 512, n, lane);
        float bn = L.ff2_b[n];
#pragma unroll
        for (int r = 0; r < 8; ++r) hs[(r + hi * 8) * HIDD + n] += acc[r] + bn;
      }
      __syncthreads();
    }

    for (int i = tid; i < BB * HIDD; i += 256)
      outs[(size_t)(t * BB) * HIDD + i] = hs[i];
    __syncthreads();
  }
}

// ---------------- heads: fused [policy(18)|baseline(1)|pad] = 32x128 f16 ------
__global__ void __launch_bounds__(256)
k_heads(const float* __restrict__ outs, const _Float16* __restrict__ hw16,
        const float* __restrict__ pb, const float* __restrict__ bb,
        float* __restrict__ logits, float* __restrict__ baseline) {
  const int lane = threadIdx.x & 31, wv = threadIdx.x >> 5;
  const int m = lane & 15, hi = lane >> 4;
  const int m0 = (blockIdx.x * 8 + wv) * 16;
  const int n  = blockIdx.y * 16 + m;
  const float* arow = outs + (size_t)(m0 + m) * HIDD;
  const _Float16* wrow = hw16 + (size_t)n * HIDD;
  v8f acc = {};
#pragma unroll
  for (int kc = 0; kc < HIDD; kc += 32) {
    float4 p0 = *(const float4*)(arow + kc + hi * 8);
    float4 p1 = *(const float4*)(arow + kc + hi * 8 + 4);
    float4 p2 = *(const float4*)(arow + kc + hi * 8 + 16);
    float4 p3 = *(const float4*)(arow + kc + hi * 8 + 20);
    v16h a;
    a[0] = (_Float16)p0.x; a[1] = (_Float16)p0.y; a[2]  = (_Float16)p0.z; a[3]  = (_Float16)p0.w;
    a[4] = (_Float16)p1.x; a[5] = (_Float16)p1.y; a[6]  = (_Float16)p1.z; a[7]  = (_Float16)p1.w;
    a[8] = (_Float16)p2.x; a[9] = (_Float16)p2.y; a[10] = (_Float16)p2.z; a[11] = (_Float16)p2.w;
    a[12] = (_Float16)p3.x; a[13] = (_Float16)p3.y; a[14] = (_Float16)p3.z; a[15] = (_Float16)p3.w;
    v16h b = load_a16(wrow + kc + hi * 8);
    acc = wmma_f16f32(a, b, acc);
  }
#pragma unroll
  for (int r = 0; r < 8; ++r) {
    int M = m0 + r + hi * 8;
    if (n < NAA)       logits[(size_t)M * NAA + n] = acc[r] + pb[n];
    else if (n == NAA) baseline[M] = acc[r] + bb[0];
  }
}

__global__ void k_argmax(const float* __restrict__ logits, float* __restrict__ action) {
  int i = blockIdx.x * blockDim.x + threadIdx.x;
  if (i >= TT * BB) return;
  const float* row = logits + (size_t)i * NAA;
  float best = row[0];
  int bi = 0;
#pragma unroll
  for (int j = 1; j < NAA; ++j)
    if (row[j] > best) { best = row[j]; bi = j; }
  action[i] = (float)bi;
}

// ---------------- host orchestration ----------------
extern "C" void kernel_launch(void* const* d_in, const int* in_sizes, int n_in,
                              void* d_out, int out_size, void* d_ws, size_t ws_size,
                              hipStream_t stream) {
  const float* frame  = (const float*)d_in[0];
  const float* reward = (const float*)d_in[1];
  const int*   done   = (const int*)d_in[2];
  const float* st0    = (const float*)d_in[3];
  const float* st1    = (const float*)d_in[4];
  const float* baseline_b = (const float*)d_in[5];
  const float* baseline_w = (const float*)d_in[6];
  auto blk = [&](int i, int j) { return (const float*)d_in[7 + i * 10 + j]; };
  const float* fc_b  = (const float*)d_in[37];
  const float* fc_w  = (const float*)d_in[38];
  const float* inp_b = (const float*)d_in[39];
  const float* inp_w = (const float*)d_in[40];
  const float* policy_b = (const float*)d_in[61];
  const float* policy_w = (const float*)d_in[62];

  char* ws = (char*)d_ws;
  size_t off = 0;
  auto take = [&](size_t bytes) -> char* {
    char* p = ws + off;
    off += (bytes + 255) & ~(size_t)255;
    return p;
  };
  _Float16* act_a = (_Float16*)take(33554432ull * 2);
  _Float16* act_b = (_Float16*)take(8388608ull * 2);
  _Float16* act_c = (_Float16*)take(8388608ull * 2);
  _Float16* act_d = (_Float16*)take(8388608ull * 2);
  _Float16* padbuf = (_Float16*)take(10653696ull * 2);
  _Float16* core_in16 = (_Float16*)take(512ull * XPITCH * 2);
  float* outs = (float*)take(512ull * 128 * 4);
  float* s0   = (float*)take(65536ull * 4);
  float* s1   = (float*)take(65536ull * 4);

  auto cvt = [&](const float* src, int rows, int K, int KPAD, int rowsPad) {
    _Float16* dst = (_Float16*)take((size_t)rowsPad * KPAD * 2);
    int total = rowsPad * KPAD;
    k_cvtpad<<<(total + 255) / 256, 256, 0, stream>>>(src, dst, rows, K, KPAD, rowsPad);
    return dst;
  };
  _Float16* cw[3][5];
  const int cco[3] = {16, 32, 32};
  const int cki[3][5] = {{27, 144, 144, 144, 144},
                         {144, 288, 288, 288, 288},
                         {288, 288, 288, 288, 288}};
  for (int bix = 0; bix < 3; ++bix)
    for (int j = 0; j < 5; ++j) {
      int K9 = cki[bix][j];
      int KP = ((K9 + 31) / 32) * 32;
      cw[bix][j] = cvt(blk(bix, j * 2 + 1), cco[bix], K9, KP, cco[bix]);
    }
  _Float16* fc_w16  = cvt(fc_w, 256, 2048, 2048, 256);
  _Float16* inp_w16 = cvt(inp_w, 128, 257, XPITCH, 128);
  CoreLayer L[2];
  for (int i = 0; i < 2; ++i) {
    const int base = 41 + i * 10;
    L[i].ff1_b  = (const float*)d_in[base + 0];
    L[i].ff2_b  = (const float*)d_in[base + 2];
    L[i].ffln_b = (const float*)d_in[base + 4];
    L[i].ffln_g = (const float*)d_in[base + 5];
    L[i].ln_b   = (const float*)d_in[base + 6];
    L[i].ln_g   = (const float*)d_in[base + 7];
    L[i].ff1_w  = cvt((const float*)d_in[base + 1], 512, 128, 128, 512);
    L[i].ff2_w  = cvt((const float*)d_in[base + 3], 128, 512, 512, 128);
    L[i].out_w  = cvt((const float*)d_in[base + 8], 128, 128, 128, 128);
    L[i].slow_w = cvt((const float*)d_in[base + 9], 388, 128, 128, 400);
  }
  _Float16* head_w16 = cvt(policy_w, 18, 128, 128, 32);
  k_cvtpad<<<1, 256, 0, stream>>>(baseline_w, head_w16 + 18 * 128, 1, 128, 128, 1);

  auto pad_f32 = [&](const float* in, int CI, int CIP, int S, float scale, int relu) {
    int total = 512 * CIP * (S + 2) * (S + 2);
    k_pad<float><<<(total + 255) / 256, 256, 0, stream>>>(in, padbuf, CI, CIP, S, scale, relu, total);
  };
  auto pad_f16 = [&](const _Float16* in, int CI, int CIP, int S, int relu) {
    int total = 512 * CIP * (S + 2) * (S + 2);
    k_pad<_Float16><<<(total + 255) / 256, 256, 0, stream>>>(in, padbuf, CI, CIP, S, 1.0f, relu, total);
  };
  auto pool = [&](const _Float16* in, _Float16* out, int C, int S) {
    int total = 512 * C * (S / 2) * (S / 2);
    k_maxpool<<<(total + 255) / 256, 256, 0, stream>>>(in, out, C, S, total);
  };
  auto convgrid = [&](int S, int CO, int NT) {
    int tiles = S * S / 16;
    int tpb = tiles < 8 ? tiles : 8;
    return dim3((unsigned)(tiles / tpb), (unsigned)(CO / (16 * NT)), 512u);
  };
  auto cblk = [&](int S) { int t = S * S / 16; return dim3(32u * (t < 8 ? t : 8)); };

  // block 0: 3 -> 16 @64, pool -> 32
  pad_f32(frame, 3, 4, 64, 1.0f / 255.0f, 0);
  k_conv3x3<3, 64, 1><<<convgrid(64, 16, 1), cblk(64), 0, stream>>>(padbuf, cw[0][0], blk(0, 0), nullptr, act_a, 16, 0);
  pool(act_a, act_b, 16, 64);
  pad_f16(act_b, 16, 18, 32, 1);
  k_conv3x3<16, 32, 1><<<convgrid(32, 16, 1), cblk(32), 0, stream>>>(padbuf, cw[0][1], blk(0, 2), nullptr, act_c, 16, 0);
  pad_f16(act_c, 16, 18, 32, 1);
  k_conv3x3<16, 32, 1><<<convgrid(32, 16, 1), cblk(32), 0, stream>>>(padbuf, cw[0][2], blk(0, 4), act_b, act_d, 16, 0);
  pad_f16(act_d, 16, 18, 32, 1);
  k_conv3x3<16, 32, 1><<<convgrid(32, 16, 1), cblk(32), 0, stream>>>(padbuf, cw[0][3], blk(0, 6), nullptr, act_c, 16, 0);
  pad_f16(act_c, 16, 18, 32, 1);
  k_conv3x3<16, 32, 1><<<convgrid(32, 16, 1), cblk(32), 0, stream>>>(padbuf, cw[0][4], blk(0, 8), act_d, act_b, 16, 0);

  // block 1: 16 -> 32 @32, pool -> 16  (NT=2: one A gather -> 2 WMMA tiles)
  pad_f16(act_b, 16, 18, 32, 0);
  k_conv3x3<16, 32, 2><<<convgrid(32, 32, 2), cblk(32), 0, stream>>>(padbuf, cw[1][0], blk(1, 0), nullptr, act_a, 32, 0);
  pool(act_a, act_d, 32, 32);
  pad_f16(act_d, 32, 32, 16, 1);
  k_conv3x3<32, 16, 2><<<convgrid(16, 32, 2), cblk(16), 0, stream>>>(padbuf, cw[1][1], blk(1, 2), nullptr, act_c, 32, 0);
  pad_f16(act_c, 32, 32, 16, 1);
  k_conv3x3<32, 16, 2><<<convgrid(16, 32, 2), cblk(16), 0, stream>>>(padbuf, cw[1][2], blk(1, 4), act_d, act_b, 32, 0);
  pad_f16(act_b, 32, 32, 16, 1);
  k_conv3x3<32, 16, 2><<<convgrid(16, 32, 2), cblk(16), 0, stream>>>(padbuf, cw[1][3], blk(1, 6), nullptr, act_c, 32, 0);
  pad_f16(act_c, 32, 32, 16, 1);
  k_conv3x3<32, 16, 2><<<convgrid(16, 32, 2), cblk(16), 0, stream>>>(padbuf, cw[1][4], blk(1, 8), act_b, act_d, 32, 0);

  // block 2: 32 -> 32 @16, pool -> 8
  pad_f16(act_d, 32, 32, 16, 0);
  k_conv3x3<32, 16, 2><<<convgrid(16, 32, 2), cblk(16), 0, stream>>>(padbuf, cw[2][0], blk(2, 0), nullptr, act_a, 32, 0);
  pool(act_a, act_b, 32, 16);
  pad_f16(act_b, 32, 32, 8, 1);
  k_conv3x3<32, 8, 2><<<convgrid(8, 32, 2), cblk(8), 0, stream>>>(padbuf, cw[2][1], blk(2, 2), nullptr, act_c, 32, 0);
  pad_f16(act_c, 32, 32, 8, 1);
  k_conv3x3<32, 8, 2><<<convgrid(8, 32, 2), cblk(8), 0, stream>>>(padbuf, cw[2][2], blk(2, 4), act_b, act_d, 32, 0);
  pad_f16(act_d, 32, 32, 8, 1);
  k_conv3x3<32, 8, 2><<<convgrid(8, 32, 2), cblk(8), 0, stream>>>(padbuf, cw[2][3], blk(2, 6), nullptr, act_c, 32, 0);
  pad_f16(act_c, 32, 32, 8, 1);
  k_conv3x3<32, 8, 2><<<convgrid(8, 32, 2), cblk(8), 0, stream>>>(padbuf, cw[2][4], blk(2, 8), act_d, act_b, 32, 1);

  // ---- FC + tail ----
  k_fc<<<dim3(4, 16), 256, 0, stream>>>(act_b, fc_w16, fc_b, core_in16);
  k_tailcols<<<(TT * BB * 32 + 255) / 256, 256, 0, stream>>>(reward, core_in16);

  // ---- recurrent core ----
  hipMemcpyAsync(s0, st0, 65536ull * 4, hipMemcpyDeviceToDevice, stream);
  hipMemcpyAsync(s1, st1, 65536ull * 4, hipMemcpyDeviceToDevice, stream);
  k_core<<<1, 256, 0, stream>>>(core_in16, done, inp_w16, inp_b, L[0], L[1], s0, s1, outs);

  // ---- heads + outputs ----
  float* out_f    = (float*)d_out;
  float* action   = out_f;
  float* logits   = out_f + 512;
  float* baseline = out_f + 512 + 512 * NAA;
  float* out_s0   = out_f + 512 + 512 * NAA + 512;
  float* out_s1   = out_s0 + 65536;
  k_heads<<<dim3(4, 2), 256, 0, stream>>>(outs, head_w16, policy_b, baseline_b, logits, baseline);
  k_argmax<<<2, 256, 0, stream>>>(logits, action);
  hipMemcpyAsync(out_s0, s0, 65536ull * 4, hipMemcpyDeviceToDevice, stream);
  hipMemcpyAsync(out_s1, s1, 65536ull * 4, hipMemcpyDeviceToDevice, stream);
}
